// S2ConvNetModified_31104153157972
// MI455X (gfx1250) — matrix operand order repaired and checked by reading, not compiled, not run on hardware
//
#include <hip/hip_runtime.h>
#include <hip/hip_bf16.h>
#include <math.h>

typedef _Float16 f16;
typedef __attribute__((ext_vector_type(8)))  _Float16 v8h;
typedef __attribute__((ext_vector_type(16))) _Float16 v16h;
typedef __attribute__((ext_vector_type(8)))  float    v8f;

__device__ __forceinline__ v16h mk16(v8h lo, v8h hi) {
  return __builtin_shufflevector(lo, hi, 0,1,2,3,4,5,6,7,8,9,10,11,12,13,14,15);
}

// ---------------------------------------------------------------------------
// Generic f32 -> f16 convert with zero padding: src [R,C] -> dst [Rp,Cp]
// ---------------------------------------------------------------------------
__global__ void k_cvt_pad(const float* __restrict__ src, f16* __restrict__ dst,
                          int R, int C, int Rp, int Cp) {
  long t = (long)blockIdx.x * blockDim.x + threadIdx.x;
  long total = (long)Rp * Cp;
  if (t >= total) return;
  int r = (int)(t / Cp);
  int c = (int)(t % Cp);
  float v = (r < R && c < C) ? src[(long)r * C + c] : 0.0f;
  dst[t] = (f16)v;
}

// ---------------------------------------------------------------------------
// c[b,cc] = sum_{beta,alpha} from_s2[cc, beta*96+alpha] * x[b,0,alpha,beta]
// ---------------------------------------------------------------------------
__global__ void k_from_s2(const float* __restrict__ x, const float* __restrict__ from_s2,
                          float* __restrict__ c) {
  int t = blockIdx.x * blockDim.x + threadIdx.x;
  if (t >= 32 * 121) return;
  int b = t / 121, cc = t % 121;
  const float* xb = x + (long)b * 96 * 96;
  const float* fr = from_s2 + (long)cc * 9216;
  float acc = 0.0f;
  for (int beta = 0; beta < 96; ++beta)
    for (int alpha = 0; alpha < 96; ++alpha)
      acc += fr[beta * 96 + alpha] * xb[alpha * 96 + beta];
  c[t] = acc;
}

// psi1[j,i] = sum_n Yk1[n,i] * w1[0,j,n] / sqrt(24)
__global__ void k_psi1(const float* __restrict__ Yk1, const float* __restrict__ w1,
                       float* __restrict__ psi1) {
  int t = blockIdx.x * blockDim.x + threadIdx.x;
  if (t >= 20 * 121) return;
  int j = t / 121, i = t % 121;
  float acc = 0.0f;
  for (int n = 0; n < 24; ++n) acc += Yk1[n * 121 + i] * w1[j * 24 + n];
  psi1[t] = acc * 0.20412414523193154f;  // 1/sqrt(24)
}

// psi2[ij,cc] = sum_n Dk2[n,cc] * w2[ij,n] / sqrt(168)
__global__ void k_psi2(const float* __restrict__ Dk2, const float* __restrict__ w2,
                       float* __restrict__ psi2) {
  int t = blockIdx.x * blockDim.x + threadIdx.x;
  if (t >= 800 * 286) return;
  int ij = t / 286, cc = t % 286;
  float acc = 0.0f;
  for (int n = 0; n < 168; ++n) acc += Dk2[n * 286 + cc] * w2[ij * 168 + n];
  psi2[t] = acc * 0.07715167498104595f;  // 1/sqrt(168)
}

// h1[(b*20+j), n*k+m (block l)] = c[b, l^2+m] * psi1[j, l^2+n]; f16, padded to 1792
__global__ void k_h1(const float* __restrict__ c, const float* __restrict__ psi1,
                     f16* __restrict__ h1) {
  int t = blockIdx.x * blockDim.x + threadIdx.x;
  if (t >= 640 * 1792) return;
  int row = t / 1792, col = t % 1792;
  if (col >= 1771) { h1[t] = (f16)0.0f; return; }
  int b = row / 20, j = row % 20;
  const int LOFF[12] = {0, 1, 10, 35, 84, 165, 286, 455, 680, 969, 1330, 1771};
  int l = 0;
  while (col >= LOFF[l + 1]) ++l;
  int k = 2 * l + 1;
  int local = col - LOFF[l];
  int n = local / k, m = local % k;
  float v = c[b * 121 + l * l + m] * psi1[j * 121 + l * l + n];
  h1[t] = (f16)v;
}

// Per-l SO(3)->SO(3) conv: h3[(b*40+j), v*k+m] = sum_{i,u} h2[(b*20+i), u*k+m] *
//                           psi2[(i*40+j), u*k+v] / sqrt(20*k);  f16, padded to 288
__global__ void k_conv2(const float* __restrict__ h2, const float* __restrict__ psi2,
                        f16* __restrict__ h3) {
  int t = blockIdx.x * blockDim.x + threadIdx.x;
  if (t >= 1280 * 288) return;
  int row = t / 288, col = t % 288;
  if (col >= 286) { h3[t] = (f16)0.0f; return; }
  int b = row / 40, j = row % 40;
  const int LOFF[7] = {0, 1, 10, 35, 84, 165, 286};
  int l = 0;
  while (col >= LOFF[l + 1]) ++l;
  int k = 2 * l + 1;
  int local = col - LOFF[l];
  int vv = local / k, m = local % k;
  float acc = 0.0f;
  for (int i = 0; i < 20; ++i) {
    const float* hb = h2 + (long)(b * 20 + i) * 288 + LOFF[l];
    const float* pb = psi2 + (long)(i * 40 + j) * 286 + LOFF[l];
    for (int u = 0; u < k; ++u) acc += hb[u * k + m] * pb[u * k + vv];
  }
  h3[t] = (f16)(acc * rsqrtf(20.0f * (float)k));
}

// out[row] = sum_g g2[row,g] * F2_from[g]
__global__ void k_out(const f16* __restrict__ g2, const float* __restrict__ F2_from,
                      float* __restrict__ out) {
  __shared__ float s[256];
  int row = blockIdx.x;
  float acc = 0.0f;
  for (int g = threadIdx.x; g < 6912; g += blockDim.x)
    acc += (float)g2[(long)row * 6912 + g] * F2_from[g];
  s[threadIdx.x] = acc;
  __syncthreads();
  for (int st = 128; st > 0; st >>= 1) {
    if ((int)threadIdx.x < st) s[threadIdx.x] += s[threadIdx.x + st];
    __syncthreads();
  }
  if (threadIdx.x == 0) out[row] = s[0];
}

// ---------------------------------------------------------------------------
// Fragment loaders (contiguous k = 16*hi+s order, same in A and B; TN / K-contiguous operands):
//   A lane L: m = L&15, 16 contiguous K at K[hi*16]   (hi = L>>4)
//   B lane L: n = L&15, 16 contiguous K at K[hi*16]
// ---------------------------------------------------------------------------
__device__ __forceinline__ v16h load_a_frag(const f16* p) {
  return *(const v16h*)p;
}
__device__ __forceinline__ v16h load_b_frag(const f16* p) {
  return *(const v16h*)p;
}

template <bool RELU, bool OUT_F16>
__device__ __forceinline__ void emit_tile(const v8f& acc, void* Out, long m, long n, int ldo) {
#pragma unroll
  for (int r = 0; r < 8; ++r) {
    float v = acc[r];
    if (RELU) v = fmaxf(v, 0.0f) * 1.4142135623730951f;  // ACT_CST * relu
    if (OUT_F16) ((f16*)Out)[(m + r) * (long)ldo + n] = (f16)v;
    else         ((float*)Out)[(m + r) * (long)ldo + n] = v;
  }
}

// ---------------------------------------------------------------------------
// TN GEMM, 32x64 wave tile (2 M-frags x 4 N-frags): 12 b128 loads : 8 WMMAs
// per 32-K step.  ~125 VGPRs statically live -> with __launch_bounds__(32,1)
// the compiler has headroom to double-buffer fragments without spilling.
// Requires M%32==0, N%64==0, K%32==0.  One wave per block (EXEC all ones).
// ---------------------------------------------------------------------------
template <bool RELU, bool OUT_F16>
__global__ void __launch_bounds__(32, 1)
k_gemm_tn_wmma_2x4(const f16* __restrict__ A, const f16* __restrict__ Bt,
                   void* __restrict__ Out, int lda, int ldb, int ldo, int K) {
  const int lane = threadIdx.x & 31;
  const int l16  = lane & 15;
  const int hi   = lane >> 4;
  const long m_base = (long)blockIdx.y * 32;
  const long n_base = (long)blockIdx.x * 64;

  v8f acc[2][4] = {};

  const f16* a0 = A + (m_base + l16) * (long)lda + hi * 16;
  const f16* a1 = a0 + 16L * lda;
  const f16* b[4];
#pragma unroll
  for (int nt = 0; nt < 4; ++nt) b[nt] = Bt + (n_base + nt * 16 + l16) * (long)ldb + hi * 16;

  for (int k0 = 0; k0 < K; k0 += 32) {
    v16h af0 = load_a_frag(a0 + k0);
    v16h af1 = load_a_frag(a1 + k0);
    v16h bf[4];
#pragma unroll
    for (int nt = 0; nt < 4; ++nt) bf[nt] = load_b_frag(b[nt] + k0);
    // stream-ahead hints for the large B operand
    __builtin_prefetch(b[0] + k0 + 256, 0, 1);
    __builtin_prefetch(b[2] + k0 + 256, 0, 1);
#pragma unroll
    for (int nt = 0; nt < 4; ++nt) {
      acc[0][nt] = __builtin_amdgcn_wmma_f32_16x16x32_f16(
          false, af0, false, bf[nt], (short)0, acc[0][nt], false, false);
      acc[1][nt] = __builtin_amdgcn_wmma_f32_16x16x32_f16(
          false, af1, false, bf[nt], (short)0, acc[1][nt], false, false);
    }
  }

#pragma unroll
  for (int mt = 0; mt < 2; ++mt)
#pragma unroll
    for (int nt = 0; nt < 4; ++nt)
      emit_tile<RELU, OUT_F16>(acc[mt][nt], Out,
                               m_base + mt * 16 + 8 * hi,
                               n_base + nt * 16 + l16, ldo);
}

// ---------------------------------------------------------------------------
// TN GEMM, 32x32 wave tile (2x2 fragments) for dims not divisible by 64.
// Requires M%32==0, N%32==0, K%32==0.
// ---------------------------------------------------------------------------
template <bool RELU, bool OUT_F16>
__global__ void __launch_bounds__(32, 1)
k_gemm_tn_wmma32(const f16* __restrict__ A, const f16* __restrict__ Bt,
                 void* __restrict__ Out, int lda, int ldb, int ldo, int K) {
  const int lane = threadIdx.x & 31;
  const int l16  = lane & 15;
  const int hi   = lane >> 4;
  const long m_base = (long)blockIdx.y * 32;
  const long n_base = (long)blockIdx.x * 32;

  v8f acc00 = {}, acc01 = {}, acc10 = {}, acc11 = {};

  const f16* a0 = A  + (m_base + l16) * (long)lda + hi * 16;
  const f16* a1 = a0 + 16L * lda;
  const f16* b0 = Bt + (n_base + l16) * (long)ldb + hi * 16;
  const f16* b1 = b0 + 16L * ldb;

  for (int k0 = 0; k0 < K; k0 += 32) {
    v16h a_lo = load_a_frag(a0 + k0);
    v16h a_hi = load_a_frag(a1 + k0);
    v16h b_lo = load_b_frag(b0 + k0);
    v16h b_hi = load_b_frag(b1 + k0);
    __builtin_prefetch(a0 + k0 + 256, 0, 1);
    __builtin_prefetch(a1 + k0 + 256, 0, 1);
    acc00 = __builtin_amdgcn_wmma_f32_16x16x32_f16(false, a_lo, false, b_lo, (short)0, acc00, false, false);
    acc01 = __builtin_amdgcn_wmma_f32_16x16x32_f16(false, a_lo, false, b_hi, (short)0, acc01, false, false);
    acc10 = __builtin_amdgcn_wmma_f32_16x16x32_f16(false, a_hi, false, b_lo, (short)0, acc10, false, false);
    acc11 = __builtin_amdgcn_wmma_f32_16x16x32_f16(false, a_hi, false, b_hi, (short)0, acc11, false, false);
  }

  emit_tile<RELU, OUT_F16>(acc00, Out, m_base +  8 * hi,      n_base + l16,      ldo);
  emit_tile<RELU, OUT_F16>(acc01, Out, m_base +  8 * hi,      n_base + 16 + l16, ldo);
  emit_tile<RELU, OUT_F16>(acc10, Out, m_base + 16 + 8 * hi,  n_base + l16,      ldo);
  emit_tile<RELU, OUT_F16>(acc11, Out, m_base + 16 + 8 * hi,  n_base + 16 + l16, ldo);
}

// ---------------------------------------------------------------------------
extern "C" void kernel_launch(void* const* d_in, const int* in_sizes, int n_in,
                              void* d_out, int out_size, void* d_ws, size_t ws_size,
                              hipStream_t stream) {
  (void)in_sizes; (void)n_in; (void)out_size; (void)ws_size;
  const float* x       = (const float*)d_in[0];
  const float* w1      = (const float*)d_in[1];
  const float* w2      = (const float*)d_in[2];
  const float* Yk1     = (const float*)d_in[3];
  const float* Dk2     = (const float*)d_in[4];
  const float* from_s2 = (const float*)d_in[5];
  const float* D1_to   = (const float*)d_in[6];
  const float* F1_from = (const float*)d_in[7];
  const float* D2_to   = (const float*)d_in[8];
  const float* F2_from = (const float*)d_in[9];
  float* out = (float*)d_out;

  char* ws = (char*)d_ws;
  size_t off = 0;
  auto alloc = [&](size_t bytes) -> void* {
    void* p = ws + off;
    off += (bytes + 255) & ~(size_t)255;
    return p;
  };
  f16*   D1h = (f16*)  alloc(32000UL * 1792 * 2);  // D1_to  f16, K padded 1771->1792
  f16*   F1h = (f16*)  alloc(  288UL * 32000 * 2); // F1_from f16, rows padded 286->288
  f16*   D2h = (f16*)  alloc( 6912UL * 288 * 2);   // D2_to  f16, K padded 286->288
  f16*   h1h = (f16*)  alloc(  640UL * 1792 * 2);  // h1 f16 padded
  f16*   g1h = (f16*)  alloc(  640UL * 32000 * 2); // relu grid 1, f16
  float* h2  = (float*)alloc(  640UL * 288 * 4);   // Fourier coeffs lmax 5, f32
  f16*   h3h = (f16*)  alloc( 1280UL * 288 * 2);   // after conv2, f16 padded
  f16*   g2h = (f16*)  alloc( 1280UL * 6912 * 2);  // relu grid 2, f16
  float* cbf = (float*)alloc(   32UL * 121 * 4);
  float* p1  = (float*)alloc(   20UL * 121 * 4);
  float* p2  = (float*)alloc(  800UL * 286 * 4);

  const int T = 256;
  auto cdiv = [](long a, long b) { return (int)((a + b - 1) / b); };

  // one-time (per call) f32->f16 conversions of the constant operands
  k_cvt_pad<<<cdiv(32000L * 1792, T), T, 0, stream>>>(D1_to, D1h, 32000, 1771, 32000, 1792);
  k_cvt_pad<<<cdiv(288L * 32000, T), T, 0, stream>>>(F1_from, F1h, 286, 32000, 288, 32000);
  k_cvt_pad<<<cdiv(6912L * 288, T), T, 0, stream>>>(D2_to, D2h, 6912, 286, 6912, 288);

  // small setup math
  k_from_s2<<<cdiv(32 * 121, 128), 128, 0, stream>>>(x, from_s2, cbf);
  k_psi1<<<cdiv(20 * 121, 128), 128, 0, stream>>>(Yk1, w1, p1);
  k_psi2<<<cdiv(800L * 286, 128), 128, 0, stream>>>(Dk2, w2, p2);
  k_h1<<<cdiv(640L * 1792, T), T, 0, stream>>>(cbf, p1, h1h);

  // GEMM 1: g1 = sqrt2*relu(h1 @ D1_to^T)   [640 x 32000], K=1792, 32x64 tiles
  k_gemm_tn_wmma_2x4<true, true><<<dim3(500, 20), 32, 0, stream>>>(h1h, D1h, g1h, 1792, 1792, 32000, 1792);

  // GEMM 2: h2 = g1 @ F1_from^T             [640 x 288],   K=32000, 32x32 tiles
  k_gemm_tn_wmma32<false, false><<<dim3(9, 20), 32, 0, stream>>>(g1h, F1h, h2, 32000, 32000, 288, 32000);

  // per-l Wigner block contraction -> h3 [1280 x 288] f16
  k_conv2<<<cdiv(1280L * 288, T), T, 0, stream>>>(h2, p2, h3h);

  // GEMM 3: g2 = sqrt2*relu(h3 @ D2_to^T)   [1280 x 6912], K=288, 32x64 tiles
  k_gemm_tn_wmma_2x4<true, true><<<dim3(108, 40), 32, 0, stream>>>(h3h, D2h, g2h, 288, 288, 6912, 288);

  // final projection to lmax 0
  k_out<<<1280, 256, 0, stream>>>(g2h, F2_from, out);
}
